// MPSLayer_1194000908507
// MI455X (gfx1250) — matrix operand order repaired and checked by reading, not compiled, not run on hardware
//
#include <hip/hip_runtime.h>
#include <stdint.h>

typedef __attribute__((ext_vector_type(16))) __bf16 v16bf;
typedef __attribute__((ext_vector_type(8)))  __bf16 v8bf;
typedef __attribute__((ext_vector_type(8)))  float  v8f;

#define DD     32
#define NSTEPS 391
#define BPW    16    // batch columns per workgroup
#define NWG    8
#define TPB    256

__global__ __launch_bounds__(TPB) void mps_chain_kernel(
    const float* __restrict__ x,          // (128, 784, 2)
    const float* __restrict__ a0,         // (32, 2)
    const float* __restrict__ cores_left, // (391, 32, 2, 32)
    const float* __restrict__ core_mid,   // (32, 10, 32)
    const float* __restrict__ cores_right,// (391, 32, 2, 32)
    const float* __restrict__ a_end,      // (32, 2)
    float* __restrict__ out)              // (128, 10)
{
  // [par][side][p][ri(row-tile)][lane][e] : WMMA A-operand, contiguous K order (k = 16*h + e)
  __shared__ __align__(32) __bf16 aop[2][2][2][2][32][16];
  // [par][side][col][k] : chain state, transposed (k contiguous per column)
  __shared__ __align__(32) __bf16 vec[2][2][BPW][DD];
  // [par][side][p][col] : per-site input weights
  __shared__ float xb[2][2][2][BPW];

  const int tid = threadIdx.x;
  const int b0  = blockIdx.x * BPW;

  // ---- initial boundary vectors -> vec[0] (2 sides * 16 cols * 32 k) ----
  for (int i = tid; i < 2 * BPW * DD; i += TPB) {
    int side = i / (BPW * DD);
    int rem  = i % (BPW * DD);
    int col  = rem / DD;
    int k    = rem % DD;
    int b    = b0 + col;
    int site = (side == 0) ? 0 : 783;
    const float* av = (side == 0) ? a0 : a_end;
    float x0 = x[(b * 784 + site) * 2 + 0];
    float x1 = x[(b * 784 + site) * 2 + 1];
    vec[0][side][col][k] = (__bf16)(av[k * 2 + 0] * x0 + av[k * 2 + 1] * x1);
  }

  float creg[16];
  float xreg = 0.f;

  // Gather step-n operands into registers (256 threads == 256 A-operand lane slots)
  auto load_step = [&](int n) {
    int lane_s = tid & 31;
    int ri     = (tid >> 5) & 1;
    int p      = (tid >> 6) & 1;
    int side   = (tid >> 7) & 1;
    int coreIdx = (side == 0) ? n : (NSTEPS - 1 - n);
    const float* core =
        ((side == 0) ? cores_left : cores_right) + coreIdx * (DD * 2 * DD);
    int M     = lane_s & 15;
    int khalf = (lane_s >> 4) ? 16 : 0;
#pragma unroll
    for (int e = 0; e < 16; ++e) {
      int K = e + khalf;   // contiguous K order: k = 16*h + e
      int src = (side == 0)
                    ? K * (2 * DD) + p * DD + (ri * 16 + M)   // A = C_p^T (left)
                    : (ri * 16 + M) * (2 * DD) + p * DD + K;  // A = C_p   (right)
      creg[e] = core[src];
    }
    if (tid < 64) {                           // x gather: 2 sides * 2 p * 16 cols
      int side2 = tid >> 5, p2 = (tid >> 4) & 1, bl = tid & 15;
      int site  = (side2 == 0) ? (1 + n) : (782 - n);
      xreg = x[((b0 + bl) * 784 + site) * 2 + p2];
    }
  };

  auto store_step = [&](int par) {
    int lane_s = tid & 31;
    int ri     = (tid >> 5) & 1;
    int p      = (tid >> 6) & 1;
    int side   = (tid >> 7) & 1;
    v16bf hv;
#pragma unroll
    for (int e = 0; e < 16; ++e) hv[e] = (__bf16)creg[e];
    *(v16bf*)&aop[par][side][p][ri][lane_s][0] = hv;   // 32B ds_store
    if (tid < 64) {
      int side2 = tid >> 5, p2 = (tid >> 4) & 1, bl = tid & 15;
      xb[par][side2][p2][bl] = xreg;
    }
  };

  load_step(0);

  for (int n = 0; n < NSTEPS; ++n) {
    const int par = n & 1;
    store_step(par);                        // phase 1: publish operands for step n
    if (n + 1 < NSTEPS) load_step(n + 1);   // phase 2: pipeline next step's gather
    if (n + 2 < NSTEPS) {                   // L2 prefetch two steps ahead
      __builtin_prefetch(cores_left  + (n + 2) * (DD * 2 * DD) + ((tid * 8) & 2047), 0, 1);
      __builtin_prefetch(cores_right + (NSTEPS - 3 - n) * (DD * 2 * DD) + ((tid * 8) & 2047), 0, 1);
    }
    __syncthreads();                        // aop[par], xb[par], vec[par] all valid

    const int w = tid >> 5;
    if (w < 4) {                            // wave-uniform: EXEC all-ones inside
      const int side = w >> 1;
      const int ri   = w & 1;
      const int lane = tid & 31;
      v16bf A0 = *(const v16bf*)&aop[par][side][0][ri][lane][0];
      v16bf A1 = *(const v16bf*)&aop[par][side][1][ri][lane][0];
      v16bf Bm = *(const v16bf*)&vec[par][side][lane & 15][(lane >> 4) * 16];
      v8f z = {};
      v8f c0 = __builtin_amdgcn_wmma_f32_16x16x32_bf16(false, A0, false, Bm,
                                                       (short)0, z, false, false);
      v8f c1 = __builtin_amdgcn_wmma_f32_16x16x32_bf16(false, A1, false, Bm,
                                                       (short)0, z, false, false);
      const float x0 = xb[par][side][0][lane & 15];
      const float x1 = xb[par][side][1][lane & 15];
      const int rowbase = ri * 16 + ((lane >> 4) ? 8 : 0);
      v8bf ov;
#pragma unroll
      for (int i = 0; i < 8; ++i) ov[i] = (__bf16)(x0 * c0[i] + x1 * c1[i]);
      *(v8bf*)&vec[par ^ 1][side][lane & 15][rowbase] = ov;   // 16B ds_store
    }
  }
  __syncthreads();

  // ---- final label contraction: out[b,c] = sum_a L[a,b] R[a,b] mid[a,c,a] ----
  const int fp = NSTEPS & 1;   // = 1
  for (int i = tid; i < BPW * 10; i += TPB) {
    int bl = i / 10, c = i % 10;
    float acc = 0.f;
#pragma unroll
    for (int a = 0; a < DD; ++a) {
      float lv = (float)vec[fp][0][bl][a];
      float rv = (float)vec[fp][1][bl][a];
      acc += lv * rv * core_mid[a * (10 * DD) + c * DD + a];
    }
    out[(b0 + bl) * 10 + c] = acc;
  }
}

extern "C" void kernel_launch(void* const* d_in, const int* in_sizes, int n_in,
                              void* d_out, int out_size, void* d_ws, size_t ws_size,
                              hipStream_t stream) {
  const float* x  = (const float*)d_in[0];
  const float* a0 = (const float*)d_in[1];
  const float* cl = (const float*)d_in[2];
  const float* cm = (const float*)d_in[3];
  const float* cr = (const float*)d_in[4];
  const float* ae = (const float*)d_in[5];
  float* out = (float*)d_out;
  hipLaunchKernelGGL(mps_chain_kernel, dim3(NWG), dim3(TPB), 0, stream,
                     x, a0, cl, cm, cr, ae, out);
}
